// LemMoEOpenEqui_85744727097790
// MI455X (gfx1250) — matrix operand order repaired and checked by reading, not compiled, not run on hardware
//
#include <hip/hip_runtime.h>
#include <cstdint>

typedef __attribute__((ext_vector_type(16))) _Float16 v16h;
typedef __attribute__((ext_vector_type(8)))  _Float16 v8h;
typedef __attribute__((ext_vector_type(8)))  float    v8f;

#define NEDGES 400000
#define NNODES 12500
#define NTILES 25000          // 400000 / 16
#define WAVES  6              // waves per block (LDS-limited)
#define THREADS (WAVES * 32)
#define AS     168            // A-tile row stride (halfs), 160 K + pad
#define WS1    168            // W1^T row stride (halfs)
#define WSX    136            // W2/W3/W_env^T row stride (halfs)
#define WGT_HALFS (128*WS1 + 3*128*WSX)   // 73728
#define WGT_BYTES (WGT_HALFS*2)           // 147456
#define OHF    2048           // floats per one-hot tile buffer (16x128)

#if __has_builtin(__builtin_amdgcn_s_wait_asynccnt)
#define WAIT_ASYNC(n) __builtin_amdgcn_s_wait_asynccnt(n)
#else
#define WAIT_ASYNC(n) asm volatile("s_wait_asynccnt %0" ::"i"(n) : "memory")
#endif

// ---------------- phase 0a: zero output (node accumulator) ----------------
__global__ void zero_out_kernel(float4* __restrict__ out, int n4) {
  int i = blockIdx.x * blockDim.x + threadIdx.x;
  int stride = gridDim.x * blockDim.x;
  float4 z = make_float4(0.f, 0.f, 0.f, 0.f);
  for (; i < n4; i += stride) out[i] = z;
}

// ---------------- phase 0b: transpose+convert weights to f16 --------------
__global__ void prep_weights_kernel(const float* __restrict__ W1,
                                    const float* __restrict__ W2,
                                    const float* __restrict__ W3,
                                    const float* __restrict__ WE,
                                    _Float16* __restrict__ ws) {
  int i = blockIdx.x * blockDim.x + threadIdx.x;
  if (i >= WGT_HALFS) return;
  if (i < 128 * WS1) {
    int n = i / WS1, k = i % WS1;                       // Wt1[n][k] = W1[k][n]
    ws[i] = (k < 136) ? (_Float16)W1[k * 128 + n] : (_Float16)0.f;
  } else {
    int j = i - 128 * WS1;
    int mat = j / (128 * WSX);
    int rem = j % (128 * WSX);
    int n = rem / WSX, k = rem % WSX;
    const float* src = (mat == 0) ? W2 : ((mat == 1) ? W3 : WE);
    ws[i] = (k < 128) ? (_Float16)src[k * 128 + n] : (_Float16)0.f;
  }
}

// ---------------- WMMA helpers (wave32 fragment layouts, ISA 7.12.2) ------
template <int KSTEPS, int WSTRIDE>
__device__ __forceinline__ void gemm_tile(const _Float16* __restrict__ A,
                                          const _Float16* __restrict__ W,
                                          int hi, int m15, v8f acc[8]) {
#pragma unroll
  for (int ks = 0; ks < KSTEPS; ++ks) {
    const int kb = ks * 32;
    // A (16xK): lane m15 = row M; elems 0..7 -> K=kb+8*hi+e ; 8..15 -> K=kb+16+8*hi+e
    v8h alo = *(const v8h*)(A + m15 * AS + kb + 8 * hi);
    v8h ahi = *(const v8h*)(A + m15 * AS + kb + 16 + 8 * hi);
    v16h a;
#pragma unroll
    for (int i = 0; i < 8; ++i) { a[i] = alo[i]; a[i + 8] = ahi[i]; }
#pragma unroll
    for (int t = 0; t < 8; ++t) {
      // B (Kx16): lane = column N, elems 0..7 -> K=kb+8*hi+e ; 8..15 -> K=kb+16+8*hi+e (W stored transposed [N][K])
      const _Float16* wp = W + (t * 16 + m15) * WSTRIDE + kb + 8 * hi;
      v8h b0 = *(const v8h*)(wp);
      v8h b1 = *(const v8h*)(wp + 16);
      v16h b;
#pragma unroll
      for (int i = 0; i < 8; ++i) { b[i] = b0[i]; b[i + 8] = b1[i]; }
      acc[t] = __builtin_amdgcn_wmma_f32_16x16x32_f16(
          false, a, false, b, (short)0, acc[t], false, false);
    }
  }
}

__device__ __forceinline__ void store_silu(_Float16* __restrict__ A, int hi,
                                           int m15, const v8f acc[8]) {
  // D f32 layout: vgpr r -> row r (lanes 0-15) / r+8 (lanes 16-31), lane&15 = col
#pragma unroll
  for (int t = 0; t < 8; ++t) {
#pragma unroll
    for (int r = 0; r < 8; ++r) {
      float v = acc[t][r];
      v = v / (1.f + __expf(-v));
      A[(r + 8 * hi) * AS + t * 16 + m15] = (_Float16)v;
    }
  }
}

// async copy of one 16x128 f32 one-hot tile into LDS (512B per instruction,
// 16 instructions, tracked on ASYNCcnt; completes in order)
__device__ __forceinline__ void oh_issue(const float* __restrict__ oh, int tile,
                                         const float* buf, int lane) {
  const char* g = (const char*)(oh + (size_t)tile * (16 * 128)) + lane * 16;
  unsigned l = (unsigned)(uintptr_t)buf + lane * 16;  // LDS byte offset
#pragma unroll
  for (int it = 0; it < 16; ++it) {
    asm volatile("global_load_async_to_lds_b128 %0, %1, off"
                 :: "v"(l + (unsigned)(it * 512)), "v"(g + (size_t)it * 512)
                 : "memory");
  }
}

// ---------------- phase 1: fused edge pipeline ----------------------------
__global__ __launch_bounds__(THREADS) void edge_kernel(
    const int* __restrict__ edge_index, const float* __restrict__ edge_sh,
    const float* __restrict__ edge_length, const float* __restrict__ edge_one_hot,
    const float* __restrict__ bessel_w, const _Float16* __restrict__ wq,
    float* __restrict__ out) {
  extern __shared__ char smem_raw[];
  _Float16* sW1 = (_Float16*)smem_raw;                 // 128 x 168
  _Float16* sW2 = sW1 + 128 * WS1;                     // 128 x 136
  _Float16* sW3 = sW2 + 128 * WSX;
  _Float16* sWE = sW3 + 128 * WSX;
  _Float16* sA  = sWE + 128 * WSX;                     // WAVES x (16 x 168)
  float* sOH  = (float*)(sA + WAVES * 16 * AS);        // WAVES x 2 x 2048 f32
  float* sSh  = sOH + WAVES * 2 * OHF;                 // WAVES x 256
  float* sCut = sSh + WAVES * 256;                     // WAVES x 16
  int*   sDst = (int*)(sCut + WAVES * 16);             // WAVES x 16

  {  // stage all 4 weight matrices into LDS once per block
    const uint4* src = (const uint4*)wq;
    uint4* dst = (uint4*)smem_raw;
    for (int i = threadIdx.x; i < WGT_BYTES / 16; i += THREADS) dst[i] = src[i];
  }
  __syncthreads();

  const int wave = threadIdx.x >> 5;
  const int lane = threadIdx.x & 31;
  const int hi = lane >> 4;
  const int m15 = lane & 15;

  _Float16* A   = sA + wave * (16 * AS);
  float* OH     = sOH + wave * (2 * OHF);
  float* shv    = sSh + wave * 256;
  float* cutv   = sCut + wave * 16;
  int*   dstv   = sDst + wave * 16;

  float bw[8];
#pragma unroll
  for (int n = 0; n < 8; ++n) bw[n] = bessel_w[n];

  const int STRIDE = gridDim.x * WAVES;
  int tile = blockIdx.x * WAVES + wave;
  int parity = 0;
  if (tile < NTILES) oh_issue(edge_one_hot, tile, OH, lane);   // prologue

  for (; tile < NTILES; tile += STRIDE, parity ^= 1) {
    const int e0 = tile * 16;
    const int nxt = tile + STRIDE;
    const bool haveNext = (nxt < NTILES);
    if (haveNext)  // issue next tile's copy behind this tile's compute
      oh_issue(edge_one_hot, nxt, OH + (parity ^ 1) * OHF, lane);

    // per-edge scalars: cutoff, bessel basis, destination node
    if (lane < 16) {
      const int e = e0 + lane;
      const float len = edge_length[e];
      const float x = len * 0.2f;                      // len / R_MAX
      const float x2 = x * x;
      const float x6 = x2 * x2 * x2;
      const float cut = 1.f - 28.f * x6 + 48.f * x6 * x - 21.f * x6 * x2;
      const float fac = (x < 1.f && cut > 0.f) ? cut : 0.f;
      cutv[lane] = fac;
      dstv[lane] = edge_index[e];                      // row 0 of edge_index
      const float invl = 1.f / len;
#pragma unroll
      for (int n = 0; n < 8; ++n) {
        float b = 0.632455532f * __sinf(len * bw[n] * 0.2f) * invl;
        A[lane * AS + 128 + n] = (_Float16)b;
      }
#pragma unroll
      for (int k = 136; k < AS; ++k) A[lane * AS + k] = (_Float16)0.f;
    }

    {  // spherical harmonics 16x16 -> LDS
      const int row = lane >> 1;
      const int c8 = (lane & 1) * 8;
      const float4 s0 = *(const float4*)(edge_sh + (size_t)(e0 + row) * 16 + c8);
      const float4 s1 = *(const float4*)(edge_sh + (size_t)(e0 + row) * 16 + c8 + 4);
      float* q = shv + row * 16 + c8;
      q[0] = s0.x; q[1] = s0.y; q[2] = s0.z; q[3] = s0.w;
      q[4] = s1.x; q[5] = s1.y; q[6] = s1.z; q[7] = s1.w;
    }

    // wait for THIS tile's async copy only (next tile's 16 remain in flight;
    // async loads complete in order -> asynccnt<=16 means ours are done)
    if (haveNext) WAIT_ASYNC(16); else WAIT_ASYNC(0);

    {  // convert staged f32 one-hot -> f16 A tile (row = it, lane = 4-col group)
      const float* cur = OH + parity * OHF;
#pragma unroll
      for (int it = 0; it < 16; ++it) {
        const float4 oh4 = *(const float4*)(cur + it * 128 + 4 * lane);
        _Float16* p = A + it * AS + 4 * lane;
        p[0] = (_Float16)oh4.x; p[1] = (_Float16)oh4.y;
        p[2] = (_Float16)oh4.z; p[3] = (_Float16)oh4.w;
      }
    }
    asm volatile("" ::: "memory");

    v8f acc[8];
    const v8f zv = {0.f, 0.f, 0.f, 0.f, 0.f, 0.f, 0.f, 0.f};

    // GEMM1: [16x160] x [160x128], silu
#pragma unroll
    for (int t = 0; t < 8; ++t) acc[t] = zv;
    gemm_tile<5, WS1>(A, sW1, hi, m15, acc);
    store_silu(A, hi, m15, acc);
    asm volatile("" ::: "memory");

    // GEMM2: [16x128] x [128x128], silu
#pragma unroll
    for (int t = 0; t < 8; ++t) acc[t] = zv;
    gemm_tile<4, WSX>(A, sW2, hi, m15, acc);
    store_silu(A, hi, m15, acc);
    asm volatile("" ::: "memory");

    // GEMM3: [16x128] x [128x128], scale rows by cutoff -> latents
#pragma unroll
    for (int t = 0; t < 8; ++t) acc[t] = zv;
    gemm_tile<4, WSX>(A, sW3, hi, m15, acc);
    {
      float fac8[8];
#pragma unroll
      for (int r = 0; r < 8; ++r) fac8[r] = cutv[r + 8 * hi];
#pragma unroll
      for (int t = 0; t < 8; ++t)
#pragma unroll
        for (int r = 0; r < 8; ++r)
          A[(r + 8 * hi) * AS + t * 16 + m15] = (_Float16)(acc[t][r] * fac8[r]);
    }
    asm volatile("" ::: "memory");

    // GEMM4: latents x W_env -> w_e [16 x 4*32]
#pragma unroll
    for (int t = 0; t < 8; ++t) acc[t] = zv;
    gemm_tile<4, WSX>(A, sWE, hi, m15, acc);

    // scatter: edge_features = w_e[l][m] * sh[s_l + j], atomically into nodes
    {
      int dst8[8];
#pragma unroll
      for (int r = 0; r < 8; ++r) dst8[r] = dstv[r + 8 * hi];
      const float scale = 0.17677669529f;  // 32^-0.5 (avg-neigh norm)
#pragma unroll
      for (int t = 0; t < 8; ++t) {
        const int l = t >> 1;                             // compile-time per t
        const int d = 2 * l + 1;
        const int off = (l == 0) ? 0 : ((l == 1) ? 32 : ((l == 2) ? 128 : 288));
        const int sbeg = (l == 0) ? 0 : ((l == 1) ? 1 : ((l == 2) ? 4 : 9));
        const int mm = (t & 1) * 16 + m15;                // mul index 0..31
#pragma unroll
        for (int r = 0; r < 8; ++r) {
          const float w = acc[t][r] * scale;
          if (w != 0.f) {
            const int row = r + 8 * hi;
            float* op = out + (size_t)dst8[r] * 512 + off + mm * d;
            const float* sp = shv + row * 16 + sbeg;
#pragma unroll
            for (int j = 0; j < d; ++j) atomicAdd(op + j, w * sp[j]);
          }
        }
      }
    }
    asm volatile("" ::: "memory");
  }
}

// ---------------- phase 2: separable layer norm (in place) ----------------
__global__ __launch_bounds__(256) void ln_kernel(float* __restrict__ out,
                                                 const float* __restrict__ ln_w,
                                                 const float* __restrict__ ln_b,
                                                 int nnodes) {
  const int node = blockIdx.x * 8 + (threadIdx.x >> 5);
  const int lane = threadIdx.x & 31;  // mul channel
  if (node >= nnodes) return;
  float* p = out + (size_t)node * 512;
  float f0 = p[lane];
  float f1[3], f2[5], f3[7];
#pragma unroll
  for (int j = 0; j < 3; ++j) f1[j] = p[32 + lane * 3 + j];
#pragma unroll
  for (int j = 0; j < 5; ++j) f2[j] = p[128 + lane * 5 + j];
#pragma unroll
  for (int j = 0; j < 7; ++j) f3[j] = p[288 + lane * 7 + j];

  float s = f0;
#pragma unroll
  for (int o = 16; o > 0; o >>= 1) s += __shfl_xor(s, o, 32);
  f0 -= s * 0.03125f;  // center scalars over mul dim

  float t1 = 0.f, t2 = 0.f, t3 = 0.f;
#pragma unroll
  for (int j = 0; j < 3; ++j) t1 += f1[j] * f1[j];
#pragma unroll
  for (int j = 0; j < 5; ++j) t2 += f2[j] * f2[j];
#pragma unroll
  for (int j = 0; j < 7; ++j) t3 += f3[j] * f3[j];
  float ss = f0 * f0 * 0.25f + t1 * (1.f / 12.f) + t2 * (1.f / 20.f) + t3 * (1.f / 28.f);
#pragma unroll
  for (int o = 16; o > 0; o >>= 1) ss += __shfl_xor(ss, o, 32);
  const float inv = rsqrtf(ss * 0.03125f + 1e-8f);

  p[lane] = f0 * inv * ln_w[lane] + ln_b[lane];
#pragma unroll
  for (int j = 0; j < 3; ++j) p[32 + lane * 3 + j] = f1[j] * inv * ln_w[32 + lane];
#pragma unroll
  for (int j = 0; j < 5; ++j) p[128 + lane * 5 + j] = f2[j] * inv * ln_w[64 + lane];
#pragma unroll
  for (int j = 0; j < 7; ++j) p[288 + lane * 7 + j] = f3[j] * inv * ln_w[96 + lane];
}

// ---------------- launcher -------------------------------------------------
extern "C" void kernel_launch(void* const* d_in, const int* in_sizes, int n_in,
                              void* d_out, int out_size, void* d_ws, size_t ws_size,
                              hipStream_t stream) {
  const int*   edge_index   = (const int*)  d_in[0];
  const float* edge_sh      = (const float*)d_in[3];
  const float* edge_length  = (const float*)d_in[4];
  const float* edge_one_hot = (const float*)d_in[5];
  const float* bessel_w     = (const float*)d_in[6];
  const float* W1           = (const float*)d_in[7];
  const float* W2           = (const float*)d_in[8];
  const float* W3           = (const float*)d_in[9];
  const float* W_env        = (const float*)d_in[10];
  const float* ln_w         = (const float*)d_in[11];
  const float* ln_b         = (const float*)d_in[12];
  float* out = (float*)d_out;
  _Float16* wq = (_Float16*)d_ws;

  zero_out_kernel<<<1024, 256, 0, stream>>>((float4*)d_out, out_size / 4);
  prep_weights_kernel<<<(WGT_HALFS + 255) / 256, 256, 0, stream>>>(W1, W2, W3, W_env, wq);

  const size_t shmem = (size_t)WGT_BYTES                 // weights
                       + WAVES * 16 * AS * 2             // A tiles (f16)
                       + WAVES * 2 * OHF * 4             // one-hot double buffers
                       + WAVES * 256 * 4                 // sh
                       + WAVES * 16 * 4                  // cutoff
                       + WAVES * 16 * 4;                 // dst  -> 284928 B
  hipFuncSetAttribute((const void*)edge_kernel,
                      hipFuncAttributeMaxDynamicSharedMemorySize, (int)shmem);
  edge_kernel<<<1024, THREADS, shmem, stream>>>(edge_index, edge_sh, edge_length,
                                                edge_one_hot, bessel_w, wq, out);

  ln_kernel<<<(NNODES + 7) / 8, 256, 0, stream>>>(out, ln_w, ln_b, NNODES);
}
